// GatedMoECrossAttn_13211319402741
// MI455X (gfx1250) — matrix operand order repaired and checked by reading, not compiled, not run on hardware
//
#include <hip/hip_runtime.h>
#include <stdint.h>

typedef __attribute__((ext_vector_type(16))) __bf16 bh16;
typedef __attribute__((ext_vector_type(8)))  float  f32x8;

constexpr int BATCH = 4;
constexpr int NT    = 2048;
constexpr int NI    = 1024;
constexpr int DM    = 1024;
constexpr int NH    = 8;
constexpr int DH    = 64;
constexpr int INNER = 512;   // NH*DH
constexpr int NE    = 8;
constexpr int HD    = 4096;
constexpr int CAP   = 512;
constexpr int NKEY  = 1056;  // 1025 keys padded to 32

union V16 { bh16 v; uint4 u[2]; };

// two 16-byte chunks -> one 16-element bf16 fragment
__device__ inline bh16 ldfrag(const __bf16* p0, const __bf16* p1) {
  V16 t;
  t.u[0] = *(const uint4*)p0;
  t.u[1] = *(const uint4*)p1;
  return t.v;
}

// CDNA5 async global->LDS copy (16B per lane), tracked by ASYNCcnt
__device__ inline void async_copy_b128(void* lds, const void* g) {
  unsigned loff = (unsigned)(uintptr_t)lds;   // low 32 bits of flat shared ptr = LDS offset
  asm volatile("global_load_async_to_lds_b128 %0, %1, off"
               :: "v"(loff), "v"(g) : "memory");
}
__device__ inline void wait_async0() {        // all async copies done
  asm volatile("s_wait_asynccnt 0" ::: "memory");
}
__device__ inline void wait_async_pending() { // current tile done, next tile (5) in flight
  asm volatile("s_wait_asynccnt 5" ::: "memory");
}

// ---------------------------------------------------------------------------
// elementwise f32 -> bf16
// ---------------------------------------------------------------------------
__global__ __launch_bounds__(256) void f2b_kernel(const float* __restrict__ x,
                                                  __bf16* __restrict__ y,
                                                  long long n) {
  long long i = (long long)blockIdx.x * 256 + threadIdx.x;
  if (i < n) y[i] = (__bf16)x[i];
}

// ---------------------------------------------------------------------------
// f32 [K][N] -> bf16 [N][K] transpose (batched via z), 32x32 LDS tiles
// ---------------------------------------------------------------------------
__global__ __launch_bounds__(256)
void transpose_b_kernel(const float* __restrict__ in, __bf16* __restrict__ out,
                        int K, int N, long long sIn, long long sOut) {
  __shared__ float tile[32][33];
  in  += (long long)blockIdx.z * sIn;
  out += (long long)blockIdx.z * sOut;
  const int n0 = blockIdx.x * 32, k0 = blockIdx.y * 32;
  const int tx = threadIdx.x & 31, ty = threadIdx.x >> 5;  // 32 x 8
#pragma unroll
  for (int i = 0; i < 4; ++i)
    tile[ty + i * 8][tx] = in[(long long)(k0 + ty + i * 8) * N + n0 + tx];
  __syncthreads();
#pragma unroll
  for (int i = 0; i < 4; ++i)
    out[(long long)(n0 + ty + i * 8) * K + k0 + tx] = (__bf16)tile[tx][ty + i * 8];
}

// ---------------------------------------------------------------------------
// row LayerNorm (gamma only) -> bf16
// ---------------------------------------------------------------------------
__global__ __launch_bounds__(256) void ln_bf16_kernel(const float* __restrict__ x,
                                                      const float* __restrict__ g,
                                                      __bf16* __restrict__ y) {
  __shared__ float red[256];
  const int row = blockIdx.x, tid = threadIdx.x;
  const float* xr = x + (long long)row * DM;
  float v[4], s = 0.f;
#pragma unroll
  for (int i = 0; i < 4; ++i) { v[i] = xr[tid + i * 256]; s += v[i]; }
  red[tid] = s; __syncthreads();
  for (int o = 128; o > 0; o >>= 1) { if (tid < o) red[tid] += red[tid + o]; __syncthreads(); }
  const float mu = red[0] / DM;
  __syncthreads();
  s = 0.f;
#pragma unroll
  for (int i = 0; i < 4; ++i) { float d = v[i] - mu; s += d * d; }
  red[tid] = s; __syncthreads();
  for (int o = 128; o > 0; o >>= 1) { if (tid < o) red[tid] += red[tid + o]; __syncthreads(); }
  const float rs = rsqrtf(red[0] / DM + 1e-5f);
#pragma unroll
  for (int i = 0; i < 4; ++i) {
    int d = tid + i * 256;
    y[(long long)row * DM + d] = (__bf16)((v[i] - mu) * rs * g[d]);
  }
}

// ---------------------------------------------------------------------------
// attended = LN(attnWo); activated = tanh(attended) + text
// ---------------------------------------------------------------------------
__global__ __launch_bounds__(256) void lntanh_kernel(const float* __restrict__ awo,
                                                     const float* __restrict__ g,
                                                     const float* __restrict__ text,
                                                     float* __restrict__ actf,
                                                     __bf16* __restrict__ actb) {
  __shared__ float red[256];
  const int row = blockIdx.x, tid = threadIdx.x;
  const float* xr = awo + (long long)row * DM;
  float v[4], s = 0.f;
#pragma unroll
  for (int i = 0; i < 4; ++i) { v[i] = xr[tid + i * 256]; s += v[i]; }
  red[tid] = s; __syncthreads();
  for (int o = 128; o > 0; o >>= 1) { if (tid < o) red[tid] += red[tid + o]; __syncthreads(); }
  const float mu = red[0] / DM;
  __syncthreads();
  s = 0.f;
#pragma unroll
  for (int i = 0; i < 4; ++i) { float d = v[i] - mu; s += d * d; }
  red[tid] = s; __syncthreads();
  for (int o = 128; o > 0; o >>= 1) { if (tid < o) red[tid] += red[tid + o]; __syncthreads(); }
  const float rs = rsqrtf(red[0] / DM + 1e-5f);
#pragma unroll
  for (int i = 0; i < 4; ++i) {
    int d = tid + i * 256;
    float nv = (v[i] - mu) * rs * g[d];
    float a = tanhf(nv) + text[(long long)row * DM + d];
    actf[(long long)row * DM + d] = a;
    actb[(long long)row * DM + d] = (__bf16)a;
  }
}

// ---------------------------------------------------------------------------
// K/V prep: kpad[b][1056][INNER] (row0 = null-k, tail zero),
//           vpadT[b][h][DH][1056] (pre-transposed V), maskadd[b][1056]
// ---------------------------------------------------------------------------
__global__ __launch_bounds__(256)
void kvprep_kernel(const __bf16* __restrict__ kvb, const float* __restrict__ nullkv,
                   const unsigned char* __restrict__ mask,
                   __bf16* __restrict__ kpad, __bf16* __restrict__ vpadT,
                   float* __restrict__ maskadd) {
  const int key = blockIdx.x;   // 0..1055
  const int bb  = blockIdx.y;
  const int tid = threadIdx.x;
#pragma unroll
  for (int i = 0; i < 2; ++i) {
    int c = tid + i * 256;      // 0..511, c = h*64 + d
    int h = c >> 6, d = c & 63;
    __bf16 kvv = (__bf16)0.f, vvv = (__bf16)0.f;
    if (key == 0) { kvv = (__bf16)nullkv[d]; vvv = (__bf16)nullkv[DH + d]; }
    else if (key <= NI) {
      kvv = kvb[(long long)(bb * NI + key - 1) * DM + c];
      vvv = kvb[(long long)(bb * NI + key - 1) * DM + INNER + c];
    }
    kpad[((long long)bb * NKEY + key) * INNER + c] = kvv;
    vpadT[((long long)(bb * NH + h) * DH + d) * NKEY + key] = vvv;
  }
  if (tid == 0) {
    bool vld = (key == 0) || (key <= NI && mask[bb * NI + key - 1] != 0);
    maskadd[(long long)bb * NKEY + key] = vld ? 0.f : -3.0e38f;
  }
}

// ---------------------------------------------------------------------------
// WMMA GEMM: C[M,N] = A[M,K](bf16) * B (bf16, pre-transposed as BT[N][K])
// block tile 64x256, K-step 32; 8 waves (2x4) -> 32x64 per wave (2x4 WMMA).
// Double-buffered global_load_async_to_lds_b128 pipeline (ASYNCcnt).
// ---------------------------------------------------------------------------
template <bool OUT_BF16, bool RELU>
__global__ __launch_bounds__(256)
void gemm_kernel(const __bf16* __restrict__ A, const __bf16* __restrict__ BT,
                 void* __restrict__ Cv, int M, int N, int K,
                 long long sA, long long sB, long long sC) {
  __shared__ __align__(16) __bf16 As[2][64][40];    // row stride 80B (16B aligned)
  __shared__ __align__(16) __bf16 Bs[2][256][40];
  const int tid = threadIdx.x, lane = tid & 31, wid = tid >> 5;
  const int wm = (wid & 1) * 32, wn = (wid >> 1) * 64;
  const int m0 = blockIdx.y * 64, n0 = blockIdx.x * 256;
  A  += (long long)blockIdx.z * sA;
  BT += (long long)blockIdx.z * sB;
  float*  Cf = (float*)Cv  + (long long)blockIdx.z * sC;
  __bf16* Ch = (__bf16*)Cv + (long long)blockIdx.z * sC;

  f32x8 acc[2][4] = {};

  const int ar = tid >> 2, ac = (tid & 3) * 8;      // 8-elem chunk per thread
  const int am = lane & 15, akb = (lane < 16) ? 0 : 8;
  const int bn = lane & 15, bkb = (lane < 16) ? 0 : 8;

  const __bf16* agsrc = A  + (long long)(m0 + ar) * K + ac;
  const __bf16* bgsrc = BT + (long long)(n0 + ar) * K + ac;
  const long long brstep = (long long)64 * K;

  auto issue_tile = [&](int k0, int buf) {
    async_copy_b128(&As[buf][ar][ac], agsrc + k0);
#pragma unroll
    for (int t = 0; t < 4; ++t)
      async_copy_b128(&Bs[buf][ar + t * 64][ac], bgsrc + t * brstep + k0);
  };

  const int ntiles = K / 32;
  issue_tile(0, 0);

  for (int i = 0; i < ntiles; ++i) {
    const int cur = i & 1;
    if (i + 1 < ntiles) {
      issue_tile((i + 1) * 32, cur ^ 1);
      wait_async_pending();          // tile i landed; tile i+1 still in flight
    } else {
      wait_async0();
    }
    __syncthreads();

    const __bf16* arow0 = &As[cur][wm + am][0];
    const __bf16* arow1 = &As[cur][wm + 16 + am][0];
    bh16 af[2], bfv[4];
    af[0] = ldfrag(arow0 + akb, arow0 + akb + 16);
    af[1] = ldfrag(arow1 + akb, arow1 + akb + 16);
#pragma unroll
    for (int ni = 0; ni < 4; ++ni) {
      const __bf16* brow = &Bs[cur][wn + 16 * ni + bn][0];
      bfv[ni] = ldfrag(brow + bkb, brow + bkb + 16);
    }
#pragma unroll
    for (int mi = 0; mi < 2; ++mi)
#pragma unroll
      for (int ni = 0; ni < 4; ++ni)
        acc[mi][ni] = __builtin_amdgcn_wmma_f32_16x16x32_bf16(
            false, af[mi], false, bfv[ni], (short)0, acc[mi][ni], false, false);
    __syncthreads();                  // buf `cur` free for tile i+2's copies
  }

  const int rhi = (lane >> 4) << 3;
  const int cn  = lane & 15;
#pragma unroll
  for (int mi = 0; mi < 2; ++mi)
#pragma unroll
    for (int ni = 0; ni < 4; ++ni)
#pragma unroll
      for (int r = 0; r < 8; ++r) {
        long long row = m0 + wm + mi * 16 + r + rhi;
        long long col = n0 + wn + ni * 16 + cn;
        float v = acc[mi][ni][r];
        if (RELU) v = v > 0.f ? v : 0.f;
        if (OUT_BF16) Ch[row * N + col] = (__bf16)v;
        else          Cf[row * N + col] = v;
      }
}

// ---------------------------------------------------------------------------
// flash attention, branchless: 1 wave per 16 queries per (b,h).
// S = QK^T (WMMA) + maskadd; online softmax; O = O*alpha + P·V (WMMA).
// ---------------------------------------------------------------------------
__global__ __launch_bounds__(32)
void attn_kernel(const __bf16* __restrict__ q, const __bf16* __restrict__ kpad,
                 const __bf16* __restrict__ vpadT, const float* __restrict__ maskadd,
                 __bf16* __restrict__ outp) {
  const int lane = threadIdx.x;
  const int q0 = blockIdx.x * 16, h = blockIdx.y, bb = blockIdx.z;

  __shared__ float Sld[16][32];
  __shared__ __align__(16) __bf16 Pld[16][32];
  __shared__ float aLds[16], lLds[16];

  const int am  = lane & 15;
  const int akb = (lane < 16) ? 0 : 8;
  const int bkb = (lane < 16) ? 0 : 8;
  const int rhi = (lane >> 4) << 3;
  const int cn  = lane & 15;

  const __bf16* qrow = q + (long long)(bb * NT + q0 + am) * INNER + h * DH;
  bh16 qa[2];
  qa[0] = ldfrag(qrow + akb,      qrow + akb + 16);
  qa[1] = ldfrag(qrow + 32 + akb, qrow + 32 + akb + 16);

  const __bf16* kbase = kpad + (long long)bb * NKEY * INNER + h * DH;
  const __bf16* vbase = vpadT + (long long)(bb * NH + h) * DH * NKEY;
  const float*  mbase = maskadd + (long long)bb * NKEY;

  float mrun = -3.0e38f, lrun = 0.f;
  f32x8 o[4] = {};

  for (int j0 = 0; j0 < NKEY; j0 += 32) {
    // ---- two 16x16 S tiles ----
#pragma unroll
    for (int st = 0; st < 2; ++st) {
      const int keyn = j0 + st * 16 + cn;
      const __bf16* kr = kbase + (long long)keyn * INNER;
      f32x8 s = {};
      s = __builtin_amdgcn_wmma_f32_16x16x32_bf16(
          false, qa[0], false, ldfrag(kr + bkb, kr + bkb + 16), (short)0, s, false, false);
      s = __builtin_amdgcn_wmma_f32_16x16x32_bf16(
          false, qa[1], false, ldfrag(kr + 32 + bkb, kr + 32 + bkb + 16), (short)0, s, false, false);
      const float madd = mbase[keyn];
#pragma unroll
      for (int r = 0; r < 8; ++r)
        Sld[r + rhi][st * 16 + cn] = s[r] * 0.125f + madd;
    }
    __syncthreads();

    // ---- online softmax (lanes 0..15, one row each) ----
    if (lane < 16) {
      float tmax = -3.4e38f;
#pragma unroll
      for (int c = 0; c < 32; ++c) tmax = fmaxf(tmax, Sld[lane][c]);
      float mnew  = fmaxf(mrun, tmax);
      float alpha = __expf(mrun - mnew);
      float ssum  = 0.f;
#pragma unroll
      for (int c = 0; c < 32; ++c) {
        float p = __expf(Sld[lane][c] - mnew);
        Pld[lane][c] = (__bf16)p;
        ssum += p;
      }
      lrun = lrun * alpha + ssum;
      mrun = mnew;
      aLds[lane] = alpha;
    }
    __syncthreads();

    // ---- rescale O, accumulate P·V ----
#pragma unroll
    for (int t = 0; t < 4; ++t)
#pragma unroll
      for (int r = 0; r < 8; ++r) o[t][r] *= aLds[r + rhi];

    const __bf16* prow = &Pld[am][0];
    const bh16 pa = ldfrag(prow + akb, prow + akb + 16);
#pragma unroll
    for (int t = 0; t < 4; ++t) {
      const __bf16* vr = vbase + (long long)(t * 16 + cn) * NKEY + j0 + bkb;
      o[t] = __builtin_amdgcn_wmma_f32_16x16x32_bf16(
          false, pa, false, ldfrag(vr, vr + 16), (short)0, o[t], false, false);
    }
    __syncthreads();
  }

  if (lane < 16) lLds[lane] = lrun;
  __syncthreads();
#pragma unroll
  for (int t = 0; t < 4; ++t)
#pragma unroll
    for (int r = 0; r < 8; ++r) {
      int row = r + rhi;
      float v = o[t][r] / lLds[row];
      outp[(long long)(bb * NT + q0 + row) * INNER + h * DH + t * 16 + cn] = (__bf16)v;
    }
}

// ---------------------------------------------------------------------------
// gate: softmax(x @ w_gate), top2, normalized gates, random-policy keep flag
// ---------------------------------------------------------------------------
__global__ __launch_bounds__(256)
void gate_kernel(const float* __restrict__ act, const float* __restrict__ wg,
                 const float* __restrict__ probs, int* __restrict__ idx1,
                 int* __restrict__ idx2, float* __restrict__ g1o,
                 float* __restrict__ g2o, int* __restrict__ keep2) {
  int t = blockIdx.x * 256 + threadIdx.x;
  if (t >= BATCH * NT) return;
  float acc[NE] = {};
  const float* xr = act + (long long)t * DM;
  for (int d = 0; d < DM; ++d) {
    float xv = xr[d];
#pragma unroll
    for (int e = 0; e < NE; ++e) acc[e] += xv * wg[d * NE + e];
  }
  float mx = acc[0];
#pragma unroll
  for (int e = 1; e < NE; ++e) mx = fmaxf(mx, acc[e]);
  float sum = 0.f;
#pragma unroll
  for (int e = 0; e < NE; ++e) { acc[e] = __expf(acc[e] - mx); sum += acc[e]; }
#pragma unroll
  for (int e = 0; e < NE; ++e) acc[e] /= sum;
  int i1 = 0; float g1 = acc[0];
#pragma unroll
  for (int e = 1; e < NE; ++e) if (acc[e] > g1) { g1 = acc[e]; i1 = e; }
  int i2 = (i1 == 0) ? 1 : 0; float g2 = acc[i2];
#pragma unroll
  for (int e = 0; e < NE; ++e)
    if (e != i1 && acc[e] > g2) { g2 = acc[e]; i2 = e; }
  float denom = g1 + g2 + 1e-9f;
  g1 /= denom; g2 /= denom;
  idx1[t] = i1; idx2[t] = i2; g1o[t] = g1; g2o[t] = g2;
  keep2[t] = (probs[t] < g2 / 0.2f) ? 1 : 0;
}

// ---------------------------------------------------------------------------
// routing: sequential exclusive-cumsum + capacity per (b,e).  <<<1,32>>>
// ---------------------------------------------------------------------------
__global__ void route_kernel(const int* __restrict__ idx1, const int* __restrict__ idx2,
                             const int* __restrict__ keep2, int* __restrict__ slot1,
                             int* __restrict__ slot2, int* __restrict__ stok) {
  int tid = threadIdx.x;
  if (tid >= 32) return;
  int bb = tid >> 3, ee = tid & 7;
  int* my = stok + (long long)(ee * BATCH + bb) * CAP;
  for (int c = 0; c < CAP; ++c) my[c] = -1;
  int cnt = 0;
  for (int n = 0; n < NT; ++n) {
    int t = bb * NT + n;
    if (idx1[t] == ee) {
      if (cnt < CAP) { slot1[t] = cnt; my[cnt] = n; }
      else slot1[t] = -1;
      cnt++;
    }
  }
  int base = cnt < CAP ? cnt : CAP;
  int c2 = 0;
  for (int n = 0; n < NT; ++n) {
    int t = bb * NT + n;
    if (idx2[t] == ee) {
      if (keep2[t]) {
        int p = base + c2;
        if (p < CAP) { slot2[t] = p; my[p] = n; }
        else slot2[t] = -1;
        c2++;
      } else slot2[t] = -1;
    }
  }
}

// ---------------------------------------------------------------------------
// dispatch gather: ein[e,b,c,:] = activated_bf16[b, token, :] (or 0)
// ---------------------------------------------------------------------------
__global__ __launch_bounds__(256)
void dispatch_kernel(const __bf16* __restrict__ actb, const int* __restrict__ stok,
                     __bf16* __restrict__ ein) {
  int slot = blockIdx.x;
  int tid = threadIdx.x;
  int bb = (slot / CAP) % BATCH;
  int n = stok[slot];
  __bf16* dst = ein + (long long)slot * DM;
  if (n < 0) {
#pragma unroll
    for (int i = 0; i < 4; ++i) dst[tid + i * 256] = (__bf16)0.f;
  } else {
    const __bf16* src = actb + (long long)(bb * NT + n) * DM;
#pragma unroll
    for (int i = 0; i < 4; ++i) dst[tid + i * 256] = src[tid + i * 256];
  }
}

// ---------------------------------------------------------------------------
// final combine: out = tanh(w1*eout[i1,slot1] + w2*eout[i2,slot2] + activated)
// ---------------------------------------------------------------------------
__global__ __launch_bounds__(256)
void combine_kernel(const float* __restrict__ actf, const __bf16* __restrict__ eout,
                    const int* __restrict__ idx1, const int* __restrict__ idx2,
                    const int* __restrict__ slot1, const int* __restrict__ slot2,
                    const float* __restrict__ g1, const float* __restrict__ g2,
                    float* __restrict__ out) {
  int t = blockIdx.x, tid = threadIdx.x;
  int bb = t / NT;
  int i1 = idx1[t], s1 = slot1[t];
  int i2 = idx2[t], s2 = slot2[t];
  float w1t = (s1 >= 0) ? g1[t] : 0.f;
  float w2t = (s2 >= 0) ? g2[t] : 0.f;
  const __bf16* e1 = eout + (long long)((i1 * BATCH + bb) * CAP + (s1 >= 0 ? s1 : 0)) * DM;
  const __bf16* e2 = eout + (long long)((i2 * BATCH + bb) * CAP + (s2 >= 0 ? s2 : 0)) * DM;
#pragma unroll
  for (int i = 0; i < 4; ++i) {
    int d = tid + i * 256;
    float v = actf[(long long)t * DM + d];
    if (s1 >= 0) v += w1t * (float)e1[d];
    if (s2 >= 0) v += w2t * (float)e2[d];
    out[(long long)t * DM + d] = tanhf(v);
  }
}

// ---------------------------------------------------------------------------
extern "C" void kernel_launch(void* const* d_in, const int* in_sizes, int n_in,
                              void* d_out, int out_size, void* d_ws, size_t ws_size,
                              hipStream_t stream) {
  const float* text  = (const float*)d_in[0];
  const float* img   = (const float*)d_in[1];
  const unsigned char* mask = (const unsigned char*)d_in[2];   // jax bool -> 1 byte
  const float* probs = (const float*)d_in[3];
  const float* ln_g  = (const float*)d_in[4];
  const float* nullkv= (const float*)d_in[5];
  const float* Wq    = (const float*)d_in[6];
  const float* Wkv   = (const float*)d_in[7];
  const float* Wo    = (const float*)d_in[8];
  const float* out_g = (const float*)d_in[9];
  const float* wg    = (const float*)d_in[10];
  const float* w1    = (const float*)d_in[11];
  const float* w2    = (const float*)d_in[12];
  float* out = (float*)d_out;

  char* base = (char*)d_ws;
  size_t off = 0;
  auto carve = [&](size_t bytes) -> void* {
    void* p = base + off;
    off += (bytes + 255) & ~(size_t)255;
    return p;
  };
  __bf16* xln   = (__bf16*)carve((size_t)BATCH * NT * DM * 2);
  __bf16* imgb  = (__bf16*)carve((size_t)BATCH * NI * DM * 2);
  __bf16* qb    = (__bf16*)carve((size_t)BATCH * NT * INNER * 2);
  __bf16* kvb   = (__bf16*)carve((size_t)BATCH * NI * DM * 2);
  __bf16* attb  = (__bf16*)carve((size_t)BATCH * NT * INNER * 2);
  float*  awo   = (float*) carve((size_t)BATCH * NT * DM * 4);
  float*  actf  = (float*) carve((size_t)BATCH * NT * DM * 4);
  __bf16* actb  = (__bf16*)carve((size_t)BATCH * NT * DM * 2);
  __bf16* WqT   = (__bf16*)carve((size_t)DM * INNER * 2);
  __bf16* WkvT  = (__bf16*)carve((size_t)DM * DM * 2);
  __bf16* WoT   = (__bf16*)carve((size_t)INNER * DM * 2);
  __bf16* w1T   = (__bf16*)carve((size_t)NE * DM * HD * 2);
  __bf16* w2T   = (__bf16*)carve((size_t)NE * HD * DM * 2);
  __bf16* kpad  = (__bf16*)carve((size_t)BATCH * NKEY * INNER * 2);
  __bf16* vpadT = (__bf16*)carve((size_t)BATCH * NH * DH * NKEY * 2);
  float*  madd  = (float*) carve((size_t)BATCH * NKEY * 4);
  int*   idx1   = (int*)   carve((size_t)BATCH * NT * 4);
  int*   idx2   = (int*)   carve((size_t)BATCH * NT * 4);
  int*   keep2  = (int*)   carve((size_t)BATCH * NT * 4);
  int*   slot1  = (int*)   carve((size_t)BATCH * NT * 4);
  int*   slot2  = (int*)   carve((size_t)BATCH * NT * 4);
  float* g1b    = (float*) carve((size_t)BATCH * NT * 4);
  float* g2b    = (float*) carve((size_t)BATCH * NT * 4);
  int*   stok   = (int*)   carve((size_t)NE * BATCH * CAP * 4);
  __bf16* ein   = (__bf16*)carve((size_t)NE * BATCH * CAP * DM * 2);
  __bf16* hid   = (__bf16*)carve((size_t)NE * BATCH * CAP * HD * 2);
  __bf16* eoutb = (__bf16*)carve((size_t)NE * BATCH * CAP * DM * 2);

  const int rows = BATCH * NT;  // 8192

  // 0. weights -> bf16 transposed [N][K]
  transpose_b_kernel<<<dim3(INNER / 32, DM / 32, 1), 256, 0, stream>>>(Wq, WqT, DM, INNER, 0, 0);
  transpose_b_kernel<<<dim3(DM / 32, DM / 32, 1), 256, 0, stream>>>(Wkv, WkvT, DM, DM, 0, 0);
  transpose_b_kernel<<<dim3(DM / 32, INNER / 32, 1), 256, 0, stream>>>(Wo, WoT, INNER, DM, 0, 0);
  transpose_b_kernel<<<dim3(HD / 32, DM / 32, NE), 256, 0, stream>>>(
      w1, w1T, DM, HD, (long long)DM * HD, (long long)HD * DM);
  transpose_b_kernel<<<dim3(DM / 32, HD / 32, NE), 256, 0, stream>>>(
      w2, w2T, HD, DM, (long long)HD * DM, (long long)DM * HD);

  // 1. LN(text) -> bf16 ; img -> bf16
  ln_bf16_kernel<<<rows, 256, 0, stream>>>(text, ln_g, xln);
  {
    long long n = (long long)BATCH * NI * DM;
    f2b_kernel<<<(unsigned)((n + 255) / 256), 256, 0, stream>>>(img, imgb, n);
  }
  // 2. projections
  gemm_kernel<true, false><<<dim3(INNER / 256, rows / 64, 1), 256, 0, stream>>>(
      xln, WqT, qb, rows, INNER, DM, 0, 0, 0);
  gemm_kernel<true, false><<<dim3(DM / 256, (BATCH * NI) / 64, 1), 256, 0, stream>>>(
      imgb, WkvT, kvb, BATCH * NI, DM, DM, 0, 0, 0);
  // 3. K/V padding + flash attention
  kvprep_kernel<<<dim3(NKEY, BATCH), 256, 0, stream>>>(kvb, nullkv, mask, kpad, vpadT, madd);
  attn_kernel<<<dim3(NT / 16, NH, BATCH), 32, 0, stream>>>(qb, kpad, vpadT, madd, attb);
  // 4. output projection -> f32, then LN + tanh + residual
  gemm_kernel<false, false><<<dim3(DM / 256, rows / 64, 1), 256, 0, stream>>>(
      attb, WoT, awo, rows, DM, INNER, 0, 0, 0);
  lntanh_kernel<<<rows, 256, 0, stream>>>(awo, out_g, text, actf, actb);
  // 5. MoE routing
  gate_kernel<<<rows / 256, 256, 0, stream>>>(actf, wg, probs, idx1, idx2, g1b, g2b, keep2);
  route_kernel<<<1, 32, 0, stream>>>(idx1, idx2, keep2, slot1, slot2, stok);
  dispatch_kernel<<<NE * BATCH * CAP, 256, 0, stream>>>(actb, stok, ein);
  // 6. expert FFN (batched over experts)
  gemm_kernel<true, true><<<dim3(HD / 256, (BATCH * CAP) / 64, NE), 256, 0, stream>>>(
      ein, w1T, hid, BATCH * CAP, HD, DM,
      (long long)BATCH * CAP * DM, (long long)DM * HD, (long long)BATCH * CAP * HD);
  gemm_kernel<true, false><<<dim3(DM / 256, (BATCH * CAP) / 64, NE), 256, 0, stream>>>(
      hid, w2T, eoutb, BATCH * CAP, DM, HD,
      (long long)BATCH * CAP * HD, (long long)HD * DM, (long long)BATCH * CAP * DM);
  // 7. combine + tanh
  combine_kernel<<<rows, 256, 0, stream>>>(actf, eoutb, idx1, idx2, slot1, slot2,
                                           g1b, g2b, out);
}
